// RpoNet2D_23845658427538
// MI455X (gfx1250) — matrix operand order repaired and checked by reading, not compiled, not run on hardware
//
#include <hip/hip_runtime.h>
#include <hip/hip_bf16.h>
#include <math.h>

// ---------- types ----------
typedef __attribute__((ext_vector_type(16))) __bf16        v16bf;
typedef __attribute__((ext_vector_type(8)))  float         v8f;
typedef __attribute__((ext_vector_type(4)))  unsigned int  v4u;
typedef __attribute__((ext_vector_type(8)))  int           v8i;
typedef __attribute__((ext_vector_type(4)))  int           v4i;

union FragBF { v16bf v; unsigned short u[16]; };

#define B_    192
#define C_    32
#define KDIM  2080   // 2048 pooled + 32 past_enc ; 65 * 32
#define HSZ   32

__device__ __forceinline__ unsigned short f2b(float f) {
    unsigned int u = __float_as_uint(f);
    u += 0x7fffu + ((u >> 16) & 1u);       // round-to-nearest-even
    return (unsigned short)(u >> 16);
}
__device__ __forceinline__ float leaky(float x) { return x > 0.f ? x : 0.01f * x; }
__device__ __forceinline__ float sigm(float x)  { return 1.f / (1.f + __expf(-x)); }

#if __has_builtin(__builtin_amdgcn_tensor_load_to_lds) && __has_builtin(__builtin_amdgcn_s_wait_tensorcnt)
#define HAVE_TDM 1
#else
#define HAVE_TDM 0
#endif

// ---------------------------------------------------------------------------
// Kernel 1: W1[:, 32:2112] f32 -> bf16, [N=64][K=2080] (B-operand source)
// ---------------------------------------------------------------------------
__global__ void convert_w1_kernel(const float* __restrict__ W1,
                                  unsigned short* __restrict__ w1b) {
    int i = blockIdx.x * blockDim.x + threadIdx.x;    // 64*2080
    if (i >= 64 * KDIM) return;
    int n = i / KDIM, k = i - n * KDIM;
    w1b[i] = f2b(W1[n * 2112 + 32 + k]);
}

// ---------------------------------------------------------------------------
// Kernel 2: past_enc = traj_past @ past_W^T + past_b -> staA[:, 2048:2080]
// ---------------------------------------------------------------------------
__global__ void past_enc_kernel(const float* __restrict__ traj_past,
                                const float* __restrict__ past_W,
                                const float* __restrict__ past_b,
                                unsigned short* __restrict__ staA) {
    int i = blockIdx.x * blockDim.x + threadIdx.x;    // 192*32
    if (i >= B_ * 32) return;
    int b = i >> 5, j = i & 31;
    float s = past_b[j];
#pragma unroll
    for (int q = 0; q < 6; ++q) s += traj_past[b * 6 + q] * past_W[j * 6 + q];
    staA[(size_t)b * KDIM + 2048 + j] = f2b(s);
}

// ---------------------------------------------------------------------------
// Kernel 3: 8x8 max-pool of feat_map via TDM tile load into LDS.
// One block per (b,c) plane (64x64 f32 = 16 KB). 64 threads.
// ---------------------------------------------------------------------------
__global__ __launch_bounds__(64) void pool_kernel(const float* __restrict__ feat_map,
                                                  unsigned short* __restrict__ staA) {
    __shared__ float plane[4096];
    int bc = blockIdx.x;                                   // b*32 + c
    const float* src = feat_map + (size_t)bc * 4096;
#if HAVE_TDM
    if (threadIdx.x < 32) {                                // TDM: one issue per block (wave 0)
        unsigned int laddr = (unsigned int)(size_t)(&plane[0]);
        unsigned long long ga = (unsigned long long)(size_t)src;
        // D# group0: count=1 | lds_addr | global_addr[56:0] | type=2 (bits 127:126)
        v4u g0 = { 1u, laddr,
                   (unsigned int)(ga & 0xFFFFFFFFu),
                   (unsigned int)(((ga >> 32) & 0x01FFFFFFu) | 0x80000000u) };
        // D# group1: data_size=4B; tensor 64x64; tile 64x64; dim0_stride=64
        v8i g1 = { (int)(2u << 16),      // data_size=2 (4 bytes), mask=0
                   (int)(64u << 16),     // tensor_dim0[15:0] in bits 63:48
                   (int)(64u << 16),     // tensor_dim0 hi=0 | tensor_dim1[15:0]
                   (int)(64u << 16),     // tensor_dim1 hi=0 | tile_dim0=64
                   64,                   // tile_dim1=64, tile_dim2=0
                   64,                   // tensor_dim0_stride lo32
                   0, 0 };
        v4i gz = { 0, 0, 0, 0 };         // groups 2/3 unused (2D tensor)
#if defined(__clang_major__) && (__clang_major__ >= 23)
        v8i gz8 = { 0, 0, 0, 0, 0, 0, 0, 0 };
        __builtin_amdgcn_tensor_load_to_lds(g0, g1, gz, gz, gz8, 0);
#else
        __builtin_amdgcn_tensor_load_to_lds(g0, g1, gz, gz, 0);
#endif
        __builtin_amdgcn_s_wait_tensorcnt(0);
    }
    __syncthreads();
#else
    for (int i = threadIdx.x; i < 4096; i += 64) plane[i] = src[i];
    __syncthreads();
#endif
    int t  = threadIdx.x;                                  // output index: zo*8+xo
    int zo = t >> 3, xo = t & 7;
    float m = -3.402823466e38f;
#pragma unroll
    for (int zi = 0; zi < 8; ++zi)
#pragma unroll
        for (int xi = 0; xi < 8; ++xi)
            m = fmaxf(m, plane[(zo * 8 + zi) * 64 + xo * 8 + xi]);
    int b = bc >> 5, c = bc & 31;
    staA[(size_t)b * KDIM + c * 64 + t] = f2b(m);
}

// ---------------------------------------------------------------------------
// Kernel 4: staC = staA(bf16) @ w1b^T + b1   (M=192, N=64, K=2080)
// One wave per 16x16 tile; 12x4 tiles = 48 blocks. bf16 WMMA, f32 accumulate.
// ---------------------------------------------------------------------------
__global__ __launch_bounds__(32) void sta_gemm_kernel(const unsigned short* __restrict__ staA,
                                                      const unsigned short* __restrict__ w1b,
                                                      const float* __restrict__ b1,
                                                      float* __restrict__ staC) {
    int tile = blockIdx.x;
    int tm = tile >> 2, tn = tile & 3;
    int lane = threadIdx.x & 31;
    int half = lane >> 4, l15 = lane & 15;

    const unsigned short* arow = staA + (size_t)(tm * 16 + l15) * KDIM;
    const unsigned short* brow = w1b + (size_t)(tn * 16 + l15) * KDIM;

    v8f acc = { 0.f, 0.f, 0.f, 0.f, 0.f, 0.f, 0.f, 0.f };
    for (int k0 = 0; k0 < KDIM; k0 += 32) {
        FragBF a, bm;
#pragma unroll
        for (int j = 0; j < 8; ++j) {                 // ISA 16-bit A layout
            a.u[j]     = arow[k0 + half * 8 + j];     // K = half*8 + 0..7
            a.u[8 + j] = arow[k0 + 16 + half * 8 + j];// K = 16 + half*8 + 0..7
        }
#pragma unroll
        for (int j = 0; j < 16; ++j)                  // B: col = l15, K = (j/8)*16 + half*8 + j%8
            bm.u[j] = brow[k0 + (j >> 3) * 16 + half * 8 + (j & 7)];
        acc = __builtin_amdgcn_wmma_f32_16x16x32_bf16(false, a.v, false, bm.v,
                                                      (short)0, acc, false, false);
        if (k0 + 32 < KDIM) {
            __builtin_prefetch(arow + k0 + 32, 0, 0); // global_prefetch_b8
            __builtin_prefetch(brow + k0 + 32, 0, 0);
        }
    }
    int n = tn * 16 + l15;
    float bias = b1[n];
#pragma unroll
    for (int v = 0; v < 8; ++v) {                     // D: M = v + 8*half, N = l15
        int m = tm * 16 + v + 8 * half;
        staC[m * 64 + n] = acc[v] + bias;
    }
}

// ---------------------------------------------------------------------------
// Kernel 5: 27-step sequential scan. One wave per 16 batch rows (12 blocks).
// Weights LDS-resident; per-step GEMMs on WMMA; tiny parts per-lane VALU.
// ---------------------------------------------------------------------------
__global__ __launch_bounds__(32) void scan_kernel(const float* __restrict__ pred_map,
                                                  const float* __restrict__ noise,
                                                  const float* __restrict__ traj_past,
                                                  const float* __restrict__ W_ih,
                                                  const float* __restrict__ W_hh,
                                                  const float* __restrict__ b_ih,
                                                  const float* __restrict__ b_hh,
                                                  const float* __restrict__ W1,
                                                  const float* __restrict__ W2,
                                                  const float* __restrict__ b2,
                                                  const float* __restrict__ W3,
                                                  const float* __restrict__ b3,
                                                  const float* __restrict__ staC,
                                                  float* __restrict__ out) {
    __shared__ float          sWih[96 * 9];
    __shared__ unsigned short sWhh[96 * 32];   // bf16, [gate][k]
    __shared__ unsigned short sW1h[64 * 32];   // bf16, W1[:, :32]
    __shared__ unsigned short sW2[32 * 64];    // bf16
    __shared__ float          sW3[6 * 32];
    __shared__ float          sbih[96], sbhh[96], sb2[32], sb3[8];
    __shared__ float          sC[16 * 64];     // sta contribution (incl. b1)
    __shared__ float          sHx[16 * 32];
    __shared__ unsigned short sHxb[16 * 32];   // bf16 hx
    __shared__ float          sGi[16 * 96];
    __shared__ float          sGh[16 * 96];
    __shared__ unsigned short sH1b[16 * 64];   // bf16 h1
    __shared__ float          sH2[16 * 32];

    const int lane = threadIdx.x & 31;
    const int half = lane >> 4, l15 = lane & 15;
    const int b0 = blockIdx.x * 16;

    // ---- stage weights / constants into LDS (once) ----
    for (int i = lane; i < 96 * 9;  i += 32) sWih[i] = W_ih[i];
    for (int i = lane; i < 96 * 32; i += 32) sWhh[i] = f2b(W_hh[i]);
    for (int i = lane; i < 64 * 32; i += 32) { int n = i >> 5, k = i & 31; sW1h[i] = f2b(W1[n * 2112 + k]); }
    for (int i = lane; i < 32 * 64; i += 32) sW2[i] = f2b(W2[i]);
    for (int i = lane; i < 6 * 32;  i += 32) sW3[i] = W3[i];
    for (int i = lane; i < 96;      i += 32) { sbih[i] = b_ih[i]; sbhh[i] = b_hh[i]; }
    if (lane < 32) sb2[lane] = b2[lane];
    if (lane < 6)  sb3[lane] = b3[lane];
    for (int i = lane; i < 16 * 64; i += 32) sC[i] = staC[(size_t)(b0 + (i >> 6)) * 64 + (i & 63)];
    for (int i = lane; i < 16 * 32; i += 32) { sHx[i] = 0.f; sHxb[i] = 0; }
    __syncthreads();

    // window state (3 points x 2 coords) in lanes 0..15
    float w0x = 0, w0z = 0, w1x = 0, w1z = 0, w2x = 0, w2z = 0;
    if (lane < 16) {
        const float* tp = traj_past + (size_t)(b0 + lane) * 6;
        w0x = tp[0]; w0z = tp[1]; w1x = tp[2]; w1z = tp[3]; w2x = tp[4]; w2z = tp[5];
    }

    for (int t = 0; t < 27; ++t) {
        // ---- (A) bilinear sample + gi (K=9, lanes 0..15, one batch row each) ----
        if (lane < 16) {
            const int b = b0 + lane;
            const float* pm = pred_map + (size_t)b * 4096;
            float xin[9];
            float wx[3] = { w0x, w1x, w2x };
            float wz[3] = { w0z, w1z, w2z };
#pragma unroll
            for (int p = 0; p < 3; ++p) {
                float xm = fminf(fmaxf((wx[p] + 16.f) * 2.f - 0.5f, 0.f), 63.f);
                float zm = fminf(fmaxf((wz[p] + 16.f) * 2.f - 0.5f, 0.f), 63.f);
                int x0 = (int)floorf(xm), z0 = (int)floorf(zm);
                int x1 = min(x0 + 1, 63), z1 = min(z0 + 1, 63);
                float fx = xm - (float)x0, fz = zm - (float)z0;
                float v00 = pm[z0 * 64 + x0], v01 = pm[z0 * 64 + x1];
                float v10 = pm[z1 * 64 + x0], v11 = pm[z1 * 64 + x1];
                xin[p] = v00 * (1.f - fx) * (1.f - fz) + v01 * fx * (1.f - fz)
                       + v10 * (1.f - fx) * fz + v11 * fx * fz;
            }
            xin[3] = w0x; xin[4] = w0z; xin[5] = w1x; xin[6] = w1z; xin[7] = w2x; xin[8] = w2z;
            for (int g = 0; g < 96; ++g) {
                float s = sbih[g];
#pragma unroll
                for (int i = 0; i < 9; ++i) s += sWih[g * 9 + i] * xin[i];
                sGi[lane * 96 + g] = s;
            }
        }
        __syncthreads();

        // ---- (B) gh = hx @ W_hh^T : 6 WMMA tiles (K=32) ----
        {
            FragBF a;
#pragma unroll
            for (int j = 0; j < 8; ++j) {
                a.u[j]     = sHxb[l15 * 32 + half * 8 + j];
                a.u[8 + j] = sHxb[l15 * 32 + 16 + half * 8 + j];
            }
#pragma unroll
            for (int tn = 0; tn < 6; ++tn) {
                FragBF bm;
                int g = tn * 16 + l15;
#pragma unroll
                for (int j = 0; j < 16; ++j) bm.u[j] = sWhh[g * 32 + (j >> 3) * 16 + half * 8 + (j & 7)];
                v8f acc = { 0.f, 0.f, 0.f, 0.f, 0.f, 0.f, 0.f, 0.f };
                acc = __builtin_amdgcn_wmma_f32_16x16x32_bf16(false, a.v, false, bm.v,
                                                              (short)0, acc, false, false);
#pragma unroll
                for (int v = 0; v < 8; ++v)
                    sGh[(v + 8 * half) * 96 + tn * 16 + l15] = acc[v];
            }
        }
        __syncthreads();

        // ---- (C) GRU pointwise: lane -> (row = l15, gates half*16..+15) ----
        {
            int bl = l15, jb = half * 16;
#pragma unroll
            for (int q = 0; q < 16; ++q) {
                int j = jb + q;
                float r = sigm(sGi[bl * 96 + j]      + sGh[bl * 96 + j]      + sbhh[j]);
                float z = sigm(sGi[bl * 96 + 32 + j] + sGh[bl * 96 + 32 + j] + sbhh[32 + j]);
                float n = tanhf(sGi[bl * 96 + 64 + j] + r * (sGh[bl * 96 + 64 + j] + sbhh[64 + j]));
                float h = (1.f - z) * n + z * sHx[bl * 32 + j];
                sHx[bl * 32 + j]  = h;
                sHxb[bl * 32 + j] = f2b(h);
            }
        }
        __syncthreads();

        // ---- (D) h1 = leaky(hx @ W1hx^T + staC) : 4 WMMA tiles (K=32) ----
        {
            FragBF a;
#pragma unroll
            for (int j = 0; j < 8; ++j) {
                a.u[j]     = sHxb[l15 * 32 + half * 8 + j];
                a.u[8 + j] = sHxb[l15 * 32 + 16 + half * 8 + j];
            }
#pragma unroll
            for (int tn = 0; tn < 4; ++tn) {
                FragBF bm;
                int n = tn * 16 + l15;
#pragma unroll
                for (int j = 0; j < 16; ++j) bm.u[j] = sW1h[n * 32 + (j >> 3) * 16 + half * 8 + (j & 7)];
                v8f acc = { 0.f, 0.f, 0.f, 0.f, 0.f, 0.f, 0.f, 0.f };
                acc = __builtin_amdgcn_wmma_f32_16x16x32_bf16(false, a.v, false, bm.v,
                                                              (short)0, acc, false, false);
#pragma unroll
                for (int v = 0; v < 8; ++v) {
                    int m = v + 8 * half;
                    float h1 = leaky(acc[v] + sC[m * 64 + n]);
                    sH1b[m * 64 + n] = f2b(h1);
                }
            }
        }
        __syncthreads();

        // ---- (E) h2 = leaky(h1 @ W2^T + b2) : 2 tiles x K=64 (2 WMMA each) ----
        {
#pragma unroll
            for (int tn = 0; tn < 2; ++tn) {
                int n = tn * 16 + l15;
                v8f acc = { 0.f, 0.f, 0.f, 0.f, 0.f, 0.f, 0.f, 0.f };
#pragma unroll
                for (int k0 = 0; k0 < 64; k0 += 32) {
                    FragBF a, bm;
#pragma unroll
                    for (int j = 0; j < 8; ++j) {
                        a.u[j]     = sH1b[l15 * 64 + k0 + half * 8 + j];
                        a.u[8 + j] = sH1b[l15 * 64 + k0 + 16 + half * 8 + j];
                    }
#pragma unroll
                    for (int j = 0; j < 16; ++j) bm.u[j] = sW2[n * 64 + k0 + (j >> 3) * 16 + half * 8 + (j & 7)];
                    acc = __builtin_amdgcn_wmma_f32_16x16x32_bf16(false, a.v, false, bm.v,
                                                                  (short)0, acc, false, false);
                }
#pragma unroll
                for (int v = 0; v < 8; ++v) {
                    int m = v + 8 * half;
                    sH2[m * 32 + n] = leaky(acc[v] + sb2[n]);
                }
            }
        }
        __syncthreads();

        // ---- (F) head + softclip + Euler update (lanes 0..15) ----
        if (lane < 16) {
            const int b = b0 + lane;
            float p[6];
#pragma unroll
            for (int j = 0; j < 6; ++j) {
                float s = sb3[j];
                for (int k = 0; k < 32; ++k) s += sH2[lane * 32 + k] * sW3[j * 32 + k];
                p[j] = s;
            }
            float s00 = p[2] + 1e-8f, s01 = p[3], s10 = p[4], s11 = p[5] + 1e-8f;
            float nrm = sqrtf(s00 * s00 + s01 * s01 + s10 * s10 + s11 * s11);
            float a = nrm * 0.2f;                       // nrm / SOFTCLIP_L
            float mx = fmaxf(a, 1.f);
            float denom = mx + __logf(__expf(a - mx) + __expf(1.f - mx));
            float inv = 1.f / denom;
            s00 *= inv; s01 *= inv; s10 *= inv; s11 *= inv;
            float sig00 = s00 * s00 + s01 * s01;
            float sig01 = s00 * s10 + s01 * s11;
            float sig10 = sig01;
            float sig11 = s10 * s10 + s11 * s11;
            float nx = noise[(size_t)(b * 27 + t) * 2 + 0];
            float nz = noise[(size_t)(b * 27 + t) * 2 + 1];
            float xdd0 = p[0] + nx * (sig00 + sig10);   // sum over i of sig[i][j]*noise[j]
            float xdd1 = p[1] + nz * (sig01 + sig11);
            float xn0 = 2.f * w2x - w1x + xdd0;
            float xn1 = 2.f * w2z - w1z + xdd1;
            out[(size_t)(b * 27 + t) * 2 + 0] = xn0;
            out[(size_t)(b * 27 + t) * 2 + 1] = xn1;
            w0x = w1x; w0z = w1z; w1x = w2x; w1z = w2z; w2x = xn0; w2z = xn1;
        }
        __syncthreads();
    }
}

// ---------------------------------------------------------------------------
// Launch
// ---------------------------------------------------------------------------
extern "C" void kernel_launch(void* const* d_in, const int* in_sizes, int n_in,
                              void* d_out, int out_size, void* d_ws, size_t ws_size,
                              hipStream_t stream) {
    const float* feat_map  = (const float*)d_in[0];
    const float* pred_map  = (const float*)d_in[1];
    const float* noise     = (const float*)d_in[2];
    const float* traj_past = (const float*)d_in[3];
    const float* W_ih      = (const float*)d_in[4];
    const float* W_hh      = (const float*)d_in[5];
    const float* b_ih      = (const float*)d_in[6];
    const float* b_hh      = (const float*)d_in[7];
    const float* past_W    = (const float*)d_in[8];
    const float* past_b    = (const float*)d_in[9];
    const float* W1        = (const float*)d_in[10];
    const float* b1        = (const float*)d_in[11];
    const float* W2        = (const float*)d_in[12];
    const float* b2        = (const float*)d_in[13];
    const float* W3        = (const float*)d_in[14];
    const float* b3        = (const float*)d_in[15];
    float* out = (float*)d_out;

    char* ws = (char*)d_ws;
    unsigned short* staA = (unsigned short*)(ws);                 // 192*2080*2 = 798720 B
    unsigned short* w1b  = (unsigned short*)(ws + 800 * 1024);    // 64*2080*2  = 266240 B
    float*          staC = (float*)(ws + 1100 * 1024);            // 192*64*4   =  49152 B

    convert_w1_kernel<<<(64 * KDIM + 255) / 256, 256, 0, stream>>>(W1, w1b);
    past_enc_kernel<<<(B_ * 32 + 255) / 256, 256, 0, stream>>>(traj_past, past_W, past_b, staA);
    pool_kernel<<<B_ * C_, 64, 0, stream>>>(feat_map, staA);
    sta_gemm_kernel<<<48, 32, 0, stream>>>(staA, w1b, b1, staC);
    scan_kernel<<<B_ / 16, 32, 0, stream>>>(pred_map, noise, traj_past,
                                            W_ih, W_hh, b_ih, b_hh,
                                            W1, W2, b2, W3, b3, staC, out);
}
